// BertCrf_22608707846904
// MI455X (gfx1250) — matrix operand order repaired and checked by reading, not compiled, not run on hardware
//
#include <hip/hip_runtime.h>
#include <hip/hip_bf16.h>
#include <math.h>

// Problem constants (from reference)
#define H    768
#define NL   12
#define NH   12
#define DH   64
#define FF   3072
#define SEQ  128
#define BATCH 16
#define NT   9
#define MTOK (BATCH*SEQ)   // 2048 token rows

typedef __bf16 bf16_t;
typedef __attribute__((ext_vector_type(16))) __bf16 v16bf;
typedef __attribute__((ext_vector_type(8)))  float  v8f;

// LDS byte-offset of a generic pointer to a __shared__ object
#define LDS_OFF(p) ((unsigned)(unsigned long long)(__attribute__((address_space(3))) char*)(p))

__device__ __forceinline__ bf16_t f2bf(float f) {
  union { float f; unsigned u; } in; in.f = f;
  unsigned r = in.u + 0x7fffu + ((in.u >> 16) & 1u);   // round-to-nearest-even
  union { unsigned short s; bf16_t b; } out; out.s = (unsigned short)(r >> 16);
  return out.b;
}

__device__ __forceinline__ float geluf(float x) {
  return 0.5f * x * (1.0f + erff(x * 0.70710678118654752440f));
}

// ---------------------------------------------------------------------------
// Generic bf16 WMMA GEMM:  out = A(bf16, MxK row-major) @ W(f32, KxN) + bias
// Block tile 128x64, K-step 32, 256 threads = 8 waves.
// Each wave owns a 32x32 C sub-tile: 2 A-frags x 2 B-frags -> 4 WMMAs/K-step.
// A tile staged via GLOBAL_LOAD_ASYNC_TO_LDS_B128 (ASYNCcnt DMA path);
// W tile converted fp32->bf16 and transposed through VGPRs.
// Optional exact-GELU epilogue; writes fp32 and/or bf16 outputs.
// ---------------------------------------------------------------------------
#define ACT_NONE 0
#define ACT_GELU 1

union Frag { v16bf v; uint4 q[2]; };

__global__ __launch_bounds__(256) void gemm_bf16_kernel(
    const bf16_t* __restrict__ A, const float* __restrict__ W,
    const float* __restrict__ bias, float* __restrict__ out32,
    bf16_t* __restrict__ out16, int M, int N, int K, int act)
{
  __shared__ __align__(16) bf16_t As[128][40];  // m x k (row-major, padded)
  __shared__ __align__(16) bf16_t Bs[64][40];   // n x k (transposed,  padded)

  const int tid  = threadIdx.x;
  const int lane = tid & 31;
  const int wave = tid >> 5;
  const int mBase = blockIdx.y * 128;
  const int nBase = blockIdx.x * 64;
  const int mSub  = (wave >> 1) * 32;   // 0,32,64,96
  const int nSub  = (wave & 1) * 32;    // 0,32

  v8f acc00 = {}, acc01 = {}, acc10 = {}, acc11 = {};

  // staging assignments
  const int am  = tid >> 1;            // 0..127 (A row)
  const int ak  = (tid & 1) * 16;      // 0,16
  const int wk  = tid >> 3;            // 0..31  (W row / k)
  const int wn0 = (tid & 7) * 8;       // 0..56

  const unsigned aLds = LDS_OFF(&As[am][ak]);   // 32B per thread, contiguous

  for (int kb = 0; kb < K; kb += 32) {
    __syncthreads();
    // A tile: async DMA global -> LDS (2 x b128 per thread), no VGPR roundtrip.
    // INST_OFFSET applies to both the LDS and the global address.
    const bf16_t* ap = &A[(size_t)(mBase + am) * K + kb + ak];
    asm volatile("global_load_async_to_lds_b128 %0, %1, off"
                 :: "v"(aLds), "v"(ap) : "memory");
    asm volatile("global_load_async_to_lds_b128 %0, %1, off offset:16"
                 :: "v"(aLds), "v"(ap) : "memory");

    // W tile: read row k (8 consecutive fp32), convert, scatter transposed.
    // This VALU work overlaps the async DMA above.
    const float* wp = &W[(size_t)(kb + wk) * N + nBase + wn0];
    const float4 w0 = *reinterpret_cast<const float4*>(wp);
    const float4 w1 = *reinterpret_cast<const float4*>(wp + 4);
    Bs[wn0 + 0][wk] = f2bf(w0.x);
    Bs[wn0 + 1][wk] = f2bf(w0.y);
    Bs[wn0 + 2][wk] = f2bf(w0.z);
    Bs[wn0 + 3][wk] = f2bf(w0.w);
    Bs[wn0 + 4][wk] = f2bf(w1.x);
    Bs[wn0 + 5][wk] = f2bf(w1.y);
    Bs[wn0 + 6][wk] = f2bf(w1.z);
    Bs[wn0 + 7][wk] = f2bf(w1.w);
    // Prefetch next K-step W panel into cache (global_prefetch_b8)
    if (kb + 32 < K) {
      __builtin_prefetch(wp + (size_t)32 * N, 0, 3);
    }
    asm volatile("s_wait_asynccnt 0x0" ::: "memory");
    __syncthreads();

    // A fragments: lane<16 -> row lane, K 0..7 & 16..23 ; lane>=16 -> K 8..15 & 24..31
    const int mRow = mSub + (lane & 15);
    const int ka   = (lane >> 4) * 8;
    Frag a0, a1;
    a0.q[0] = *reinterpret_cast<const uint4*>(&As[mRow][ka]);
    a0.q[1] = *reinterpret_cast<const uint4*>(&As[mRow][ka + 16]);
    a1.q[0] = *reinterpret_cast<const uint4*>(&As[mRow + 16][ka]);
    a1.q[1] = *reinterpret_cast<const uint4*>(&As[mRow + 16][ka + 16]);

    // B fragments: lane<16 -> col lane, K 0..7 & 16..23 ; lane>=16 -> K 8..15 & 24..31
    const int kbf = (lane >> 4) * 8;
    const int c0  = nSub + (lane & 15);
    Frag b0, b1;
    b0.q[0] = *reinterpret_cast<const uint4*>(&Bs[c0][kbf]);
    b0.q[1] = *reinterpret_cast<const uint4*>(&Bs[c0][kbf + 16]);
    b1.q[0] = *reinterpret_cast<const uint4*>(&Bs[c0 + 16][kbf]);
    b1.q[1] = *reinterpret_cast<const uint4*>(&Bs[c0 + 16][kbf + 16]);

    acc00 = __builtin_amdgcn_wmma_f32_16x16x32_bf16(false, a0.v, false, b0.v,
                                                    (short)0, acc00, false, false);
    acc01 = __builtin_amdgcn_wmma_f32_16x16x32_bf16(false, a0.v, false, b1.v,
                                                    (short)0, acc01, false, false);
    acc10 = __builtin_amdgcn_wmma_f32_16x16x32_bf16(false, a1.v, false, b0.v,
                                                    (short)0, acc10, false, false);
    acc11 = __builtin_amdgcn_wmma_f32_16x16x32_bf16(false, a1.v, false, b1.v,
                                                    (short)0, acc11, false, false);
  }

  // Epilogue: C layout -> VGPR r: lanes0-15 (M=r, N=lane), lanes16-31 (M=r+8, N=lane-16)
  const int row0 = mBase + mSub + ((lane >> 4) ? 8 : 0);
  const int row1 = row0 + 16;
  const int colA = nBase + nSub + (lane & 15);
  const int colB = colA + 16;
  const float bA = bias[colA];
  const float bB = bias[colB];
  #pragma unroll
  for (int r = 0; r < 8; ++r) {
    float v00 = acc00[r] + bA;
    float v01 = acc01[r] + bB;
    float v10 = acc10[r] + bA;
    float v11 = acc11[r] + bB;
    if (act == ACT_GELU) {
      v00 = geluf(v00); v01 = geluf(v01);
      v10 = geluf(v10); v11 = geluf(v11);
    }
    const size_t i00 = (size_t)(row0 + r) * N + colA;
    const size_t i01 = (size_t)(row0 + r) * N + colB;
    const size_t i10 = (size_t)(row1 + r) * N + colA;
    const size_t i11 = (size_t)(row1 + r) * N + colB;
    if (out32) {
      out32[i00] = v00; out32[i01] = v01;
      out32[i10] = v10; out32[i11] = v11;
    }
    if (out16) {
      out16[i00] = f2bf(v00); out16[i01] = f2bf(v01);
      out16[i10] = f2bf(v10); out16[i11] = f2bf(v11);
    }
  }
}

// ---------------------------------------------------------------------------
// Embedding gather + LayerNorm (one block per token row)
// ---------------------------------------------------------------------------
__global__ __launch_bounds__(256) void embed_ln_kernel(
    const int* __restrict__ ids, const int* __restrict__ tti,
    const float* __restrict__ wemb, const float* __restrict__ pemb,
    const float* __restrict__ temb, const float* __restrict__ g,
    const float* __restrict__ bta, float* __restrict__ h32,
    bf16_t* __restrict__ h16)
{
  const int r = blockIdx.x;
  const int s = r % SEQ;
  const int t = threadIdx.x;
  __shared__ float red[256];
  const size_t wbase = (size_t)ids[r] * H;
  const size_t tbase = (size_t)tti[r] * H;
  float x[3], sum = 0.f, sq = 0.f;
  #pragma unroll
  for (int i = 0; i < 3; ++i) {
    const int c = t + i * 256;
    const float v = wemb[wbase + c] + pemb[(size_t)s * H + c] + temb[tbase + c];
    x[i] = v; sum += v; sq += v * v;
  }
  red[t] = sum; __syncthreads();
  for (int st = 128; st > 0; st >>= 1) { if (t < st) red[t] += red[t + st]; __syncthreads(); }
  const float mu = red[0] / (float)H; __syncthreads();
  red[t] = sq; __syncthreads();
  for (int st = 128; st > 0; st >>= 1) { if (t < st) red[t] += red[t + st]; __syncthreads(); }
  const float var  = red[0] / (float)H - mu * mu;
  const float rstd = rsqrtf(var + 1e-12f);
  #pragma unroll
  for (int i = 0; i < 3; ++i) {
    const int c = t + i * 256;
    const float y = g[c] * (x[i] - mu) * rstd + bta[c];
    h32[(size_t)r * H + c] = y;
    h16[(size_t)r * H + c] = f2bf(y);
  }
}

// ---------------------------------------------------------------------------
// Residual add + LayerNorm (one block per token row)
// ---------------------------------------------------------------------------
__global__ __launch_bounds__(256) void resid_ln_kernel(
    float* __restrict__ h32, bf16_t* __restrict__ h16,
    const float* __restrict__ delta, const float* __restrict__ g,
    const float* __restrict__ bta)
{
  const int r = blockIdx.x;
  const int t = threadIdx.x;
  __shared__ float red[256];
  float x[3], sum = 0.f, sq = 0.f;
  #pragma unroll
  for (int i = 0; i < 3; ++i) {
    const int c = t + i * 256;
    const float v = h32[(size_t)r * H + c] + delta[(size_t)r * H + c];
    x[i] = v; sum += v; sq += v * v;
  }
  red[t] = sum; __syncthreads();
  for (int st = 128; st > 0; st >>= 1) { if (t < st) red[t] += red[t + st]; __syncthreads(); }
  const float mu = red[0] / (float)H; __syncthreads();
  red[t] = sq; __syncthreads();
  for (int st = 128; st > 0; st >>= 1) { if (t < st) red[t] += red[t + st]; __syncthreads(); }
  const float var  = red[0] / (float)H - mu * mu;
  const float rstd = rsqrtf(var + 1e-12f);
  #pragma unroll
  for (int i = 0; i < 3; ++i) {
    const int c = t + i * 256;
    const float y = g[c] * (x[i] - mu) * rstd + bta[c];
    h32[(size_t)r * H + c] = y;
    h16[(size_t)r * H + c] = f2bf(y);
  }
}

// ---------------------------------------------------------------------------
// Attention: one block per (batch, head), one thread per query row.
// Online softmax; K/V chunks staged to LDS via async DMA. ctx written as bf16.
// ---------------------------------------------------------------------------
__global__ __launch_bounds__(128) void attn_kernel(
    const float* __restrict__ q, const float* __restrict__ k,
    const float* __restrict__ v, const int* __restrict__ mask,
    bf16_t* __restrict__ ctx16)
{
  const int bh = blockIdx.x;
  const int b  = bh / NH;
  const int hd = bh % NH;
  const int r  = threadIdx.x;             // query row 0..127
  __shared__ __align__(16) float Ks[32][64];
  __shared__ __align__(16) float Vs[32][64];
  __shared__ float mbias[32];

  float qr[64];
  const float* qp = &q[((size_t)(b * SEQ + r)) * H + hd * DH];
  #pragma unroll
  for (int d = 0; d < 64; d += 4) {
    const float4 tq = *reinterpret_cast<const float4*>(&qp[d]);
    qr[d] = tq.x; qr[d + 1] = tq.y; qr[d + 2] = tq.z; qr[d + 3] = tq.w;
  }
  float mx = -3.0e38f, l = 0.f;
  float o[64];
  #pragma unroll
  for (int d = 0; d < 64; ++d) o[d] = 0.f;

  const int jrow = threadIdx.x >> 2;         // 0..31
  const int seg  = (threadIdx.x & 3) * 16;   // 0..48
  const unsigned kLds = LDS_OFF(&Ks[jrow][seg]);   // 64B per thread, contiguous
  const unsigned vLds = LDS_OFF(&Vs[jrow][seg]);

  for (int j0 = 0; j0 < SEQ; j0 += 32) {
    __syncthreads();
    const float* kp = &k[((size_t)(b * SEQ + j0 + jrow)) * H + hd * DH + seg];
    const float* vp = &v[((size_t)(b * SEQ + j0 + jrow)) * H + hd * DH + seg];
    asm volatile("global_load_async_to_lds_b128 %0, %1, off"
                 :: "v"(kLds), "v"(kp) : "memory");
    asm volatile("global_load_async_to_lds_b128 %0, %1, off offset:16"
                 :: "v"(kLds), "v"(kp) : "memory");
    asm volatile("global_load_async_to_lds_b128 %0, %1, off offset:32"
                 :: "v"(kLds), "v"(kp) : "memory");
    asm volatile("global_load_async_to_lds_b128 %0, %1, off offset:48"
                 :: "v"(kLds), "v"(kp) : "memory");
    asm volatile("global_load_async_to_lds_b128 %0, %1, off"
                 :: "v"(vLds), "v"(vp) : "memory");
    asm volatile("global_load_async_to_lds_b128 %0, %1, off offset:16"
                 :: "v"(vLds), "v"(vp) : "memory");
    asm volatile("global_load_async_to_lds_b128 %0, %1, off offset:32"
                 :: "v"(vLds), "v"(vp) : "memory");
    asm volatile("global_load_async_to_lds_b128 %0, %1, off offset:48"
                 :: "v"(vLds), "v"(vp) : "memory");
    if (threadIdx.x < 32)
      mbias[threadIdx.x] =
          (1.0f - (float)mask[b * SEQ + j0 + threadIdx.x]) * -1.0e9f;
    asm volatile("s_wait_asynccnt 0x0" ::: "memory");
    __syncthreads();

    for (int jj = 0; jj < 32; ++jj) {
      float s = 0.f;
      #pragma unroll
      for (int d = 0; d < 64; ++d) s += qr[d] * Ks[jj][d];
      s = s * 0.125f + mbias[jj];
      const float nm    = fmaxf(mx, s);
      const float scale = __expf(mx - nm);
      const float p     = __expf(s - nm);
      l = l * scale + p;
      #pragma unroll
      for (int d = 0; d < 64; ++d) o[d] = o[d] * scale + p * Vs[jj][d];
      mx = nm;
    }
  }

  const float inv = 1.0f / l;
  bf16_t* cp = &ctx16[((size_t)(b * SEQ + r)) * H + hd * DH];
  #pragma unroll
  for (int d = 0; d < 64; ++d) cp[d] = f2bf(o[d] * inv);
}

// ---------------------------------------------------------------------------
// Classifier head: emissions[r][j] = h @ fc_w(768x9) + fc_b  (one block/row)
// ---------------------------------------------------------------------------
__global__ __launch_bounds__(256) void fc_kernel(
    const float* __restrict__ h32, const float* __restrict__ fcw,
    const float* __restrict__ fcb, float* __restrict__ out)
{
  const int r = blockIdx.x;
  float acc[NT];
  #pragma unroll
  for (int j = 0; j < NT; ++j) acc[j] = 0.f;
  for (int kk = threadIdx.x; kk < H; kk += 256) {
    const float hv = h32[(size_t)r * H + kk];
    #pragma unroll
    for (int j = 0; j < NT; ++j) acc[j] += hv * fcw[kk * NT + j];
  }
  __shared__ float red[256];
  for (int j = 0; j < NT; ++j) {
    red[threadIdx.x] = acc[j]; __syncthreads();
    for (int st = 128; st > 0; st >>= 1) {
      if (threadIdx.x < st) red[threadIdx.x] += red[threadIdx.x + st];
      __syncthreads();
    }
    if (threadIdx.x == 0) out[(size_t)r * NT + j] = red[0] + fcb[j];
    __syncthreads();
  }
}

// ---------------------------------------------------------------------------
// CRF negative log-likelihood: one wave, one batch per lane (lanes 0..15)
// ---------------------------------------------------------------------------
__global__ __launch_bounds__(32) void crf_kernel(
    const float* __restrict__ em, const int* __restrict__ tags,
    const int* __restrict__ mask, const float* __restrict__ cstart,
    const float* __restrict__ cend, const float* __restrict__ ctrans,
    float* __restrict__ lossOut)
{
  const int b = threadIdx.x;
  __shared__ float sred[32];
  float nll = 0.f;
  if (b < BATCH) {
    const float* E  = &em[(size_t)b * SEQ * NT];
    const int*   tg = &tags[b * SEQ];
    const int*   mk = &mask[b * SEQ];

    // numerator
    const int t0 = tg[0];
    float num = cstart[t0] + E[t0];
    int msum = mk[0];
    for (int s = 1; s < SEQ; ++s) {
      const float mf = (float)mk[s];
      num += (ctrans[tg[s - 1] * NT + tg[s]] + E[s * NT + tg[s]]) * mf;
      msum += mk[s];
    }
    num += cend[tg[msum - 1]];

    // partition function (alpha recursion)
    float alpha[NT];
    #pragma unroll
    for (int j = 0; j < NT; ++j) alpha[j] = cstart[j] + E[j];
    for (int s = 1; s < SEQ; ++s) {
      if (mk[s]) {
        float na[NT];
        #pragma unroll
        for (int j = 0; j < NT; ++j) {
          float m = -3.0e38f;
          #pragma unroll
          for (int i = 0; i < NT; ++i)
            m = fmaxf(m, alpha[i] + ctrans[i * NT + j]);
          float sm = 0.f;
          #pragma unroll
          for (int i = 0; i < NT; ++i)
            sm += __expf(alpha[i] + ctrans[i * NT + j] - m);
          na[j] = m + __logf(sm) + E[s * NT + j];
        }
        #pragma unroll
        for (int j = 0; j < NT; ++j) alpha[j] = na[j];
      }
    }
    float m = -3.0e38f;
    #pragma unroll
    for (int j = 0; j < NT; ++j) m = fmaxf(m, alpha[j] + cend[j]);
    float sm = 0.f;
    #pragma unroll
    for (int j = 0; j < NT; ++j) sm += __expf(alpha[j] + cend[j] - m);
    const float logZ = m + __logf(sm);
    nll = logZ - num;
  }
  sred[threadIdx.x] = nll;
  __syncthreads();
  for (int st = 16; st > 0; st >>= 1) {
    if (threadIdx.x < st) sred[threadIdx.x] += sred[threadIdx.x + st];
    __syncthreads();
  }
  if (threadIdx.x == 0) lossOut[0] = sred[0] / (float)BATCH;
}

// ---------------------------------------------------------------------------
// Host-side orchestration
// ---------------------------------------------------------------------------
extern "C" void kernel_launch(void* const* d_in, const int* in_sizes, int n_in,
                              void* d_out, int out_size, void* d_ws, size_t ws_size,
                              hipStream_t stream) {
  (void)in_sizes; (void)n_in; (void)out_size; (void)ws_size;
  const int*   ids      = (const int*)  d_in[0];
  const int*   mask     = (const int*)  d_in[1];
  const int*   tti      = (const int*)  d_in[2];
  const int*   tags     = (const int*)  d_in[3];
  const float* word_emb = (const float*)d_in[4];
  const float* pos_emb  = (const float*)d_in[5];
  const float* type_emb = (const float*)d_in[6];
  const float* emb_g    = (const float*)d_in[7];
  const float* emb_b    = (const float*)d_in[8];
  const float* Wq       = (const float*)d_in[9];
  const float* bq       = (const float*)d_in[10];
  const float* Wk       = (const float*)d_in[11];
  const float* bk       = (const float*)d_in[12];
  const float* Wv       = (const float*)d_in[13];
  const float* bv       = (const float*)d_in[14];
  const float* Wo       = (const float*)d_in[15];
  const float* bo       = (const float*)d_in[16];
  const float* ln1_g    = (const float*)d_in[17];
  const float* ln1_b    = (const float*)d_in[18];
  const float* Wi       = (const float*)d_in[19];
  const float* bi       = (const float*)d_in[20];
  const float* Wo2      = (const float*)d_in[21];
  const float* bo2      = (const float*)d_in[22];
  const float* ln2_g    = (const float*)d_in[23];
  const float* ln2_b    = (const float*)d_in[24];
  const float* fc_w     = (const float*)d_in[25];
  const float* fc_b     = (const float*)d_in[26];
  const float* crf_s    = (const float*)d_in[27];
  const float* crf_e    = (const float*)d_in[28];
  const float* crf_t    = (const float*)d_in[29];

  char* ws = (char*)d_ws;
  float*  h32    = (float*) (ws + 0);            //  6 MB
  float*  tmp32  = (float*) (ws + 6291456);      //  6 MB
  float*  q32    = (float*) (ws + 12582912);     //  6 MB
  float*  k32    = (float*) (ws + 18874368);     //  6 MB
  float*  v32    = (float*) (ws + 25165824);     //  6 MB
  bf16_t* h16    = (bf16_t*)(ws + 31457280);     //  3 MB
  bf16_t* ctx16  = (bf16_t*)(ws + 34603008);     //  3 MB
  bf16_t* ffn16  = (bf16_t*)(ws + 37748736);     // 12 MB  (end @ 48 MB)

  float* emissions = (float*)d_out;              // [2048 x 9]
  float* loss      = emissions + (size_t)MTOK * NT;

  embed_ln_kernel<<<MTOK, 256, 0, stream>>>(ids, tti, word_emb, pos_emb,
                                            type_emb, emb_g, emb_b, h32, h16);

  const dim3 gH(H / 64, MTOK / 128);    // 12 x 16
  const dim3 gF(FF / 64, MTOK / 128);   // 48 x 16

  for (int l = 0; l < NL; ++l) {
    const float* wq = Wq + (size_t)l * H * H;
    const float* wk = Wk + (size_t)l * H * H;
    const float* wv = Wv + (size_t)l * H * H;
    const float* wo = Wo + (size_t)l * H * H;
    const float* wi = Wi + (size_t)l * H * FF;
    const float* w2 = Wo2 + (size_t)l * FF * H;

    gemm_bf16_kernel<<<gH, 256, 0, stream>>>(h16, wq, bq + l * H, q32, nullptr,
                                             MTOK, H, H, ACT_NONE);
    gemm_bf16_kernel<<<gH, 256, 0, stream>>>(h16, wk, bk + l * H, k32, nullptr,
                                             MTOK, H, H, ACT_NONE);
    gemm_bf16_kernel<<<gH, 256, 0, stream>>>(h16, wv, bv + l * H, v32, nullptr,
                                             MTOK, H, H, ACT_NONE);
    attn_kernel<<<BATCH * NH, 128, 0, stream>>>(q32, k32, v32, mask, ctx16);
    gemm_bf16_kernel<<<gH, 256, 0, stream>>>(ctx16, wo, bo + l * H, tmp32,
                                             nullptr, MTOK, H, H, ACT_NONE);
    resid_ln_kernel<<<MTOK, 256, 0, stream>>>(h32, h16, tmp32, ln1_g + l * H,
                                              ln1_b + l * H);
    gemm_bf16_kernel<<<gF, 256, 0, stream>>>(h16, wi, bi + l * FF, nullptr,
                                             ffn16, MTOK, FF, H, ACT_GELU);
    gemm_bf16_kernel<<<gH, 256, 0, stream>>>(ffn16, w2, bo2 + l * H, tmp32,
                                             nullptr, MTOK, H, FF, ACT_NONE);
    resid_ln_kernel<<<MTOK, 256, 0, stream>>>(h32, h16, tmp32, ln2_g + l * H,
                                              ln2_b + l * H);
  }

  fc_kernel<<<MTOK, 256, 0, stream>>>(h32, fc_w, fc_b, emissions);
  crf_kernel<<<1, 32, 0, stream>>>(emissions, tags, mask, crf_s, crf_e, crf_t,
                                   loss);
}
